// MultiHeadAttention_38654705664734
// MI455X (gfx1250) — matrix operand order repaired and checked by reading, not compiled, not run on hardware
//
#include <hip/hip_runtime.h>
#include <hip/hip_bf16.h>

// MHA forward for B=2, S=2048, D=1024, H=16, DK=64 (fp32 reference).
// Strategy: bf16 WMMA (v_wmma_f32_16x16x32_bf16) with fp32 accumulation.
// GEMM waves use 64x64 tiles: 16 WMMA per 16 b128 loads (A/B each reused 4x).

typedef __bf16 bf16_t;
typedef bf16_t v16bf __attribute__((ext_vector_type(16)));
typedef float  v8f   __attribute__((ext_vector_type(8)));

#define NEG_BIG (-3.0e38f)

__device__ __forceinline__ unsigned short f2bf(float f) {
  unsigned int u = __float_as_uint(f);
  u += 0x7FFFu + ((u >> 16) & 1u);   // round to nearest even
  return (unsigned short)(u >> 16);
}

union FragU { v16bf v; uint4 u[2]; };

// A-fragment (16x32 bf16, MxK), src row-major [M][K] with leading dim ld.
// ISA layout: lane l holds row M=l%16; elems 0..7 -> K = 8*(l/16)+e,
// elems 8..15 -> K = 16 + 8*(l/16) + (e-8).
__device__ __forceinline__ v16bf load_a_frag(const unsigned short* src, int ld, int lane) {
  const int row = lane & 15;
  const int kb  = (lane >> 4) << 3;        // 0 or 8
  FragU t;
  t.u[0] = *(const uint4*)(src + (size_t)row * ld + kb);
  t.u[1] = *(const uint4*)(src + (size_t)row * ld + kb + 16);
  return t.v;
}

// B-fragment (32x16 bf16, KxN), src stored N-major [N][K] with leading dim ld
// (i.e. K-contiguous per column). ISA layout: lane l holds column N=l%16;
// elems 0..7 -> K = 8*(l/16) + e, elems 8..15 -> K = 16 + 8*(l/16) + (e-8).
__device__ __forceinline__ v16bf load_b_frag(const unsigned short* src, int ld, int lane) {
  const int col = lane & 15;
  const int kb  = (lane >> 4) << 3;        // 0 or 8
  FragU t;
  t.u[0] = *(const uint4*)(src + (size_t)col * ld + kb);
  t.u[1] = *(const uint4*)(src + (size_t)col * ld + kb + 16);
  return t.v;
}

__device__ __forceinline__ v8f wmma_bf16(v16bf a, v16bf b, v8f c) {
  return __builtin_amdgcn_wmma_f32_16x16x32_bf16(false, a, false, b, (short)0, c, false, false);
}

// ---------------------------------------------------------------------------
// fp32 -> bf16 bulk convert (4 elements / thread)
// ---------------------------------------------------------------------------
__global__ void cvt_f32_bf16(const float* __restrict__ src,
                             unsigned short* __restrict__ dst, int n4) {
  int i = blockIdx.x * blockDim.x + threadIdx.x;
  if (i >= n4) return;
  float4 f = ((const float4*)src)[i];
  uint2 o;
  o.x = (unsigned int)f2bf(f.x) | ((unsigned int)f2bf(f.y) << 16);
  o.y = (unsigned int)f2bf(f.z) | ((unsigned int)f2bf(f.w) << 16);
  ((uint2*)dst)[i] = o;
}

// ---------------------------------------------------------------------------
// QKV projection: Y = X @ W^T + b, M=4096, N=1024, K=1024.
// blockIdx.z: 0->q [B,H,S,DK], 1->k [B,H,S,DK], 2->v transposed [B,H,DK,S]
// 256 threads = 8 waves laid out 4(m) x 2(n); block tile 256x128;
// each wave computes a 64x64 tile = 4x4 C fragments, 16 WMMA per K-step.
// ---------------------------------------------------------------------------
__global__ __launch_bounds__(256, 1) void qkv_proj(
    const unsigned short* __restrict__ Qb, const unsigned short* __restrict__ Kb,
    const unsigned short* __restrict__ Vb,
    const unsigned short* __restrict__ Wqb, const unsigned short* __restrict__ Wkb,
    const unsigned short* __restrict__ Wvb,
    const float* __restrict__ bq, const float* __restrict__ bk,
    const float* __restrict__ bv,
    unsigned short* __restrict__ qd, unsigned short* __restrict__ kd,
    unsigned short* __restrict__ vTd) {
  const int z = blockIdx.z;
  const unsigned short* X    = (z == 0) ? Qb  : (z == 1) ? Kb  : Vb;
  const unsigned short* W    = (z == 0) ? Wqb : (z == 1) ? Wkb : Wvb;
  const float*          bias = (z == 0) ? bq  : (z == 1) ? bk  : bv;
  unsigned short*       dst  = (z == 0) ? qd  : (z == 1) ? kd  : vTd;

  const int lane = threadIdx.x & 31;
  const int wave = threadIdx.x >> 5;
  const int m0 = blockIdx.x * 256 + (wave >> 1) * 64;
  const int n0 = blockIdx.y * 128 + (wave & 1) * 64;

  v8f c[4][4] = {};

  for (int k0 = 0; k0 < 1024; k0 += 32) {
    v16bf a[4], b[4];
#pragma unroll
    for (int i = 0; i < 4; i++)
      a[i] = load_a_frag(X + (size_t)(m0 + i * 16) * 1024 + k0, 1024, lane);
#pragma unroll
    for (int j = 0; j < 4; j++)
      b[j] = load_b_frag(W + (size_t)(n0 + j * 16) * 1024 + k0, 1024, lane);
    // unconditional speculative prefetch of the next A K-slab (one line/lane)
    __builtin_prefetch(X + (size_t)(m0 + lane) * 1024 + k0 + 32, 0, 0);
#pragma unroll
    for (int i = 0; i < 4; i++)
#pragma unroll
      for (int j = 0; j < 4; j++)
        c[i][j] = wmma_bf16(a[i], b[j], c[i][j]);
  }

  const int ln = lane & 15, lh = lane >> 4;
#pragma unroll
  for (int j = 0; j < 4; j++) {
    const int n = n0 + j * 16 + ln;
    const float bval = bias[n];
    const int h = n >> 6, dk = n & 63;
#pragma unroll
    for (int i = 0; i < 4; i++) {
#pragma unroll
      for (int r = 0; r < 8; r++) {
        const int m = m0 + i * 16 + r + 8 * lh;     // token index
        const int bi = m >> 11, s = m & 2047;
        const float val = c[i][j][r] + bval;
        size_t addr;
        if (z < 2)
          addr = (((size_t)(bi * 16 + h) * 2048 + s) * 64 + dk);
        else
          addr = (((size_t)(bi * 16 + h) * 64 + dk) * 2048 + s);
        dst[addr] = f2bf(val);
      }
    }
  }
}

// ---------------------------------------------------------------------------
// Flash attention. grid = (B*H, S/64). 128 threads = 4 waves,
// each wave owns 16 query rows. K-tiles of 64 keys, online softmax.
// q,k: [B,H,S,DK] bf16; vT: [B,H,DK,S] bf16; ctx out: [B*S, D] bf16.
// ---------------------------------------------------------------------------
__global__ __launch_bounds__(128, 1) void attn(
    const unsigned short* __restrict__ q, const unsigned short* __restrict__ k,
    const unsigned short* __restrict__ vT, const int* __restrict__ mask,
    unsigned short* __restrict__ ctx) {
  __shared__ __align__(16) unsigned short Pt[4][16 * 64];

  const int lane = threadIdx.x & 31;
  const int wave = threadIdx.x >> 5;
  const int bh = blockIdx.x;          // b*16 + h
  const int bi = bh >> 4;
  const int h  = bh & 15;
  const int qr0 = blockIdx.y * 64 + wave * 16;

  const unsigned short* qbh = q  + (size_t)bh * 2048 * 64;
  const unsigned short* kbh = k  + (size_t)bh * 2048 * 64;
  const unsigned short* vbh = vT + (size_t)bh * 64 * 2048;

  // Q fragments for this wave's 16 rows (16x64 -> 2 A fragments), kept live.
  v16bf qa[2];
#pragma unroll
  for (int kc = 0; kc < 2; kc++)
    qa[kc] = load_a_frag(qbh + (size_t)qr0 * 64 + kc * 32, 64, lane);

  v8f o[4] = {};
  float mi[8], li[8];
#pragma unroll
  for (int j = 0; j < 8; j++) { mi[j] = NEG_BIG; li[j] = 0.f; }

  const int ln = lane & 15, lh = lane >> 4;
  const float scale = 0.125f;  // 1/sqrt(64)

  for (int kt0 = 0; kt0 < 2048; kt0 += 64) {
    // ---- scores S = (Q K^T) * scale, 16x64 in 4 C fragments ----
    v8f s[4] = {};
#pragma unroll
    for (int nf = 0; nf < 4; nf++) {
#pragma unroll
      for (int kc = 0; kc < 2; kc++) {
        v16bf kb = load_b_frag(kbh + (size_t)(kt0 + nf * 16) * 64 + kc * 32, 64, lane);
        s[nf] = wmma_bf16(qa[kc], kb, s[nf]);
      }
    }
#pragma unroll
    for (int nf = 0; nf < 4; nf++) {
      const int key = kt0 + nf * 16 + ln;
      const bool valid = mask[bi * 2048 + key] != 0;
#pragma unroll
      for (int j = 0; j < 8; j++) {
        const float v = s[nf][j] * scale;
        s[nf][j] = valid ? v : NEG_BIG;
      }
    }
    // ---- row max (reduce over 64 keys: 4 frags elementwise + 16 lanes) ----
    float mloc[8];
#pragma unroll
    for (int j = 0; j < 8; j++)
      mloc[j] = fmaxf(fmaxf(s[0][j], s[1][j]), fmaxf(s[2][j], s[3][j]));
#pragma unroll
    for (int off = 1; off < 16; off <<= 1)
#pragma unroll
      for (int j = 0; j < 8; j++)
        mloc[j] = fmaxf(mloc[j], __shfl_xor(mloc[j], off, 32));

    float alpha[8], rs[8];
#pragma unroll
    for (int j = 0; j < 8; j++) {
      const float mn = fmaxf(mi[j], mloc[j]);
      alpha[j] = __expf(mi[j] - mn);
      mi[j] = mn;
      rs[j] = 0.f;
    }
    // ---- P = exp(S - m); row sums; stage P in LDS (C-layout -> row-major) --
#pragma unroll
    for (int nf = 0; nf < 4; nf++) {
#pragma unroll
      for (int j = 0; j < 8; j++) {
        const float p = __expf(s[nf][j] - mi[j]);
        rs[j] += p;
        Pt[wave][(j + 8 * lh) * 64 + nf * 16 + ln] = f2bf(p);
      }
    }
#pragma unroll
    for (int off = 1; off < 16; off <<= 1)
#pragma unroll
      for (int j = 0; j < 8; j++) rs[j] += __shfl_xor(rs[j], off, 32);
#pragma unroll
    for (int j = 0; j < 8; j++) li[j] = li[j] * alpha[j] + rs[j];
    // ---- rescale running O, then O += P @ V ----
#pragma unroll
    for (int nf = 0; nf < 4; nf++)
#pragma unroll
      for (int j = 0; j < 8; j++) o[nf][j] *= alpha[j];
#pragma unroll
    for (int kc = 0; kc < 2; kc++) {
      v16bf pa = load_a_frag(&Pt[wave][0] + kc * 32, 64, lane);
#pragma unroll
      for (int nf = 0; nf < 4; nf++) {
        v16bf vb = load_b_frag(vbh + (size_t)(nf * 16) * 2048 + kt0 + kc * 32, 2048, lane);
        o[nf] = wmma_bf16(pa, vb, o[nf]);
      }
    }
  }

  // ---- normalize and store ctx[b*S+s][h*64+dk] as bf16 ----
#pragma unroll
  for (int j = 0; j < 8; j++) li[j] = 1.0f / li[j];
#pragma unroll
  for (int nf = 0; nf < 4; nf++) {
    const int dk = nf * 16 + ln;
#pragma unroll
    for (int j = 0; j < 8; j++) {
      const int srow = qr0 + j + 8 * lh;
      const size_t m = (size_t)bi * 2048 + srow;
      ctx[m * 1024 + h * 64 + dk] = f2bf(o[nf][j] * li[j]);
    }
  }
}

// ---------------------------------------------------------------------------
// Output projection: out = ctx @ Wo^T + bo (fp32 out). Same tiling as qkv_proj.
// ---------------------------------------------------------------------------
__global__ __launch_bounds__(256, 1) void out_proj(
    const unsigned short* __restrict__ ctx, const unsigned short* __restrict__ Wob,
    const float* __restrict__ bo, float* __restrict__ out) {
  const int lane = threadIdx.x & 31;
  const int wave = threadIdx.x >> 5;
  const int m0 = blockIdx.x * 256 + (wave >> 1) * 64;
  const int n0 = blockIdx.y * 128 + (wave & 1) * 64;

  v8f c[4][4] = {};
  for (int k0 = 0; k0 < 1024; k0 += 32) {
    v16bf a[4], b[4];
#pragma unroll
    for (int i = 0; i < 4; i++)
      a[i] = load_a_frag(ctx + (size_t)(m0 + i * 16) * 1024 + k0, 1024, lane);
#pragma unroll
    for (int j = 0; j < 4; j++)
      b[j] = load_b_frag(Wob + (size_t)(n0 + j * 16) * 1024 + k0, 1024, lane);
    __builtin_prefetch(ctx + (size_t)(m0 + lane) * 1024 + k0 + 32, 0, 0);
#pragma unroll
    for (int i = 0; i < 4; i++)
#pragma unroll
      for (int j = 0; j < 4; j++)
        c[i][j] = wmma_bf16(a[i], b[j], c[i][j]);
  }

  const int ln = lane & 15, lh = lane >> 4;
#pragma unroll
  for (int j = 0; j < 4; j++) {
    const int n = n0 + j * 16 + ln;
    const float bval = bo[n];
#pragma unroll
    for (int i = 0; i < 4; i++)
#pragma unroll
      for (int r = 0; r < 8; r++) {
        const int m = m0 + i * 16 + r + 8 * lh;
        out[(size_t)m * 1024 + n] = c[i][j][r] + bval;
      }
  }
}

// ---------------------------------------------------------------------------
extern "C" void kernel_launch(void* const* d_in, const int* in_sizes, int n_in,
                              void* d_out, int out_size, void* d_ws, size_t ws_size,
                              hipStream_t stream) {
  const float* Q    = (const float*)d_in[0];
  const float* K    = (const float*)d_in[1];
  const float* V    = (const float*)d_in[2];
  const int*   mask = (const int*)d_in[3];
  const float* Wq = (const float*)d_in[4];  const float* bq = (const float*)d_in[5];
  const float* Wk = (const float*)d_in[6];  const float* bk = (const float*)d_in[7];
  const float* Wv = (const float*)d_in[8];  const float* bv = (const float*)d_in[9];
  const float* Wo = (const float*)d_in[10]; const float* bo = (const float*)d_in[11];
  float* out = (float*)d_out;

  const size_t NX = (size_t)4096 * 1024;  // activation elements (B*S*D)
  const size_t NW = (size_t)1024 * 1024;  // weight elements (D*D)
  unsigned short* wsu = (unsigned short*)d_ws;
  unsigned short* Qb  = wsu;            // bf16 copies of inputs
  unsigned short* Kb  = Qb  + NX;
  unsigned short* Vb  = Kb  + NX;
  unsigned short* Wqb = Vb  + NX;       // bf16 weights
  unsigned short* Wkb = Wqb + NW;
  unsigned short* Wvb = Wkb + NW;
  unsigned short* Wob = Wvb + NW;
  unsigned short* qd  = Wob + NW;       // q  [B,H,S,DK]
  unsigned short* kd  = qd  + NX;       // k  [B,H,S,DK]
  unsigned short* vTd = kd  + NX;       // vT [B,H,DK,S]
  unsigned short* ctx = vTd + NX;       // attention context [B*S, D]
  (void)ws_size; (void)in_sizes; (void)n_in; (void)out_size;
  // total ws use: (7*NX + 4*NW) * 2 bytes = 64 MB

  const int n4x = (int)(NX / 4), n4w = (int)(NW / 4);
  cvt_f32_bf16<<<(n4x + 255) / 256, 256, 0, stream>>>(Q,  Qb,  n4x);
  cvt_f32_bf16<<<(n4x + 255) / 256, 256, 0, stream>>>(K,  Kb,  n4x);
  cvt_f32_bf16<<<(n4x + 255) / 256, 256, 0, stream>>>(V,  Vb,  n4x);
  cvt_f32_bf16<<<(n4w + 255) / 256, 256, 0, stream>>>(Wq, Wqb, n4w);
  cvt_f32_bf16<<<(n4w + 255) / 256, 256, 0, stream>>>(Wk, Wkb, n4w);
  cvt_f32_bf16<<<(n4w + 255) / 256, 256, 0, stream>>>(Wv, Wvb, n4w);
  cvt_f32_bf16<<<(n4w + 255) / 256, 256, 0, stream>>>(Wo, Wob, n4w);

  qkv_proj<<<dim3(16, 8, 3), 256, 0, stream>>>(Qb, Kb, Vb, Wqb, Wkb, Wvb,
                                               bq, bk, bv, qd, kd, vTd);
  attn<<<dim3(32, 32), 128, 0, stream>>>(qd, kd, vTd, mask, ctx);
  out_proj<<<dim3(16, 8), 256, 0, stream>>>(ctx, Wob, bo, out);
}
